// BELayer_18622978196207
// MI455X (gfx1250) — matrix operand order repaired and checked by reading, not compiled, not run on hardware
//
#include <hip/hip_runtime.h>
#include <math.h>

typedef __bf16 bf16_t;
typedef __attribute__((ext_vector_type(16))) __bf16 v16bf;
typedef __attribute__((ext_vector_type(8)))  __bf16 v8bf;
typedef __attribute__((ext_vector_type(8)))  float  v8f;

#define BATCH 16384
#define NBR   8
#define VOCAB 256
#define FFN   1024
#define MT    64      // rows per workgroup (4 row-groups x 16 rows, 2 waves share a group)
#define FC    32      // FFN chunk = WMMA K for stage 2
#define NCHUNK (FFN / FC)
#define LN_EPS 1e-5f

__device__ __forceinline__ float gelu_exact(float v) {
    return 0.5f * v * (1.0f + erff(v * 0.70710678118654752f));
}

// 16 bf16 (32B contiguous) -> v16bf fragment (two b128 loads)
__device__ __forceinline__ v16bf load_bf16x16(const bf16_t* p) {
    v8bf lo = *reinterpret_cast<const v8bf*>(p);
    v8bf hi = *reinterpret_cast<const v8bf*>(p + 8);
    return __builtin_shufflevector(lo, hi, 0,1,2,3,4,5,6,7,8,9,10,11,12,13,14,15);
}

// ---- prep: convert + transpose weights to bf16, K-contiguous ----
// U1 (8,256,1024) f32 -> U1T (8,1024,256) bf16 : U1T[br][f][d] = U1[br][d][f]
__global__ void cvt_u1(const float* __restrict__ U1, bf16_t* __restrict__ U1T) {
    size_t i = (size_t)blockIdx.x * blockDim.x + threadIdx.x;   // 8*1024*256
    int d = i & (VOCAB - 1);
    size_t t = i >> 8;
    int f = t & (FFN - 1);
    int br = (int)(t >> 10);
    U1T[i] = (bf16_t)U1[((size_t)br * VOCAB + d) * FFN + f];
}
// U2 (8,1024,256) f32 -> U2T (8,256,1024) bf16 : U2T[br][d][f] = U2[br][f][d]
__global__ void cvt_u2(const float* __restrict__ U2, bf16_t* __restrict__ U2T) {
    size_t i = (size_t)blockIdx.x * blockDim.x + threadIdx.x;   // 8*256*1024
    int f = i & (FFN - 1);
    size_t t = i >> 10;
    int d = t & (VOCAB - 1);
    int br = (int)(t >> 8);
    U2T[i] = (bf16_t)U2[((size_t)br * FFN + f) * VOCAB + d];
}

// ---- fused branch-MLP: x@U1+b1 -> gelu -> @U2+b2 -> LN -> gelu -> +skip ----
__global__ __launch_bounds__(256)
void fused_branch_mlp(const float* __restrict__ x,
                      const bf16_t* __restrict__ U1T,
                      const float* __restrict__ b1,
                      const bf16_t* __restrict__ U2T,
                      const float* __restrict__ b2,
                      const float* __restrict__ lnw,
                      const float* __restrict__ lnb,
                      float* __restrict__ out) {
    const int br   = blockIdx.y;
    const int m0   = blockIdx.x * MT;
    const int tid  = threadIdx.x;
    const int lane = tid & 31;
    const int wave = tid >> 5;
    const int half = lane >> 4;     // 0: lanes 0-15, 1: lanes 16-31
    const int ln16 = lane & 15;
    const int g    = wave & 3;      // row group: rows m0+16g .. m0+16g+15
    const int c    = wave >> 2;     // output column half: d in [c*128, c*128+128)

    __shared__ alignas(16) bf16_t sU2[2][VOCAB][40]; // double-buffered U2 chunk (80B stride)
    __shared__ alignas(16) bf16_t sH[MT][40];        // gelu'd h chunk (80B stride)
    __shared__ float sLN[2][MT][2];                  // per-half LN partials {sum, sumsq}

    const bf16_t* u1base = U1T + (size_t)br * FFN * VOCAB;
    const bf16_t* u2base = U2T + (size_t)br * VOCAB * FFN;

    // cooperative async copy of one U2 chunk (256 rows x 32 bf16) into sU2[buf]
    // thread t copies row t (64B) as 4 x async b128 into the padded 80B LDS row.
    auto issue_u2_copy = [&](int fcn, int buf) {
        const bf16_t* src = u2base + (size_t)tid * FFN + fcn * FC;
        uint32_t la = (uint32_t)(uintptr_t)&sU2[buf][tid][0];
        uint64_t ga = (uint64_t)(uintptr_t)src;
#pragma unroll
        for (int q = 0; q < 4; ++q) {
            asm volatile("global_load_async_to_lds_b128 %0, %1, off"
                         :: "v"(la + q * 16u), "v"(ga + (uint64_t)(q * 16))
                         : "memory");
        }
    };

    // ---- preload this lane's x A-fragments (row = mrow, K=0..255 as 8 fragments) ----
    const int mrow = m0 + g * 16 + ln16;
    const float* xrow = x + (size_t)mrow * (NBR * VOCAB) + br * VOCAB;
    v16bf xfrag[8];
#pragma unroll
    for (int ks = 0; ks < 8; ++ks) {
        v8f a0 = *reinterpret_cast<const v8f*>(xrow + ks * 32 + half * 16);
        v8f a1 = *reinterpret_cast<const v8f*>(xrow + ks * 32 + half * 16 + 8);
        v16bf f;
#pragma unroll
        for (int j = 0; j < 8; ++j) { f[j] = (bf16_t)a0[j]; f[8 + j] = (bf16_t)a1[j]; }
        xfrag[ks] = f;
    }

    // ---- 8 output accumulator tiles (16 rows x 128 cols per wave) ----
    v8f acc[8];
#pragma unroll
    for (int dt = 0; dt < 8; ++dt) acc[dt] = (v8f){0.f,0.f,0.f,0.f,0.f,0.f,0.f,0.f};

    issue_u2_copy(0, 0);   // prologue: DMA chunk 0 into buffer 0

    for (int fcn = 0; fcn < NCHUNK; ++fcn) {
        const int f0  = fcn * FC;
        const int cur = fcn & 1;

        // kick off DMA of the next U2 chunk; it overlaps stage-1 compute below
        if (fcn + 1 < NCHUNK) issue_u2_copy(fcn + 1, cur ^ 1);

        // ---- stage 1: this wave computes one 16x16 h tile (cols f0 + c*16 + ln16) ----
        const bf16_t* brow = u1base + (size_t)(f0 + c * 16 + ln16) * VOCAB;
        v8f h1 = (v8f){0.f,0.f,0.f,0.f,0.f,0.f,0.f,0.f};
        v16bf bcur = load_bf16x16(brow + half * 16);   // ks = 0
#pragma unroll
        for (int ks = 0; ks < 8; ++ks) {
            v16bf bnext = bcur;
            if (ks < 7) bnext = load_bf16x16(brow + (ks + 1) * 32 + half * 16);
            h1 = __builtin_amdgcn_wmma_f32_16x16x32_bf16(false, xfrag[ks], false, bcur,
                                                         (short)0, h1, false, false);
            bcur = bnext;
        }

        // ---- bias + exact gelu, write bf16 to LDS (C-layout -> row-major) ----
        {
            float bias = b1[br * FFN + f0 + c * 16 + ln16];
#pragma unroll
            for (int r = 0; r < 8; ++r) {
                float v = gelu_exact(h1[r] + bias);
                sH[g * 16 + half * 8 + r][c * 16 + ln16] = (bf16_t)v;
            }
        }

        // current U2 buffer (older 4 async ops) must be complete; async loads finish in order
        if (fcn + 1 < NCHUNK) asm volatile("s_wait_asynccnt 0x4" ::: "memory");
        else                  asm volatile("s_wait_asynccnt 0x0" ::: "memory");
        __syncthreads();   // covers both sH and sU2[cur] visibility

        // ---- stage 2: acc += gelu(h) @ U2[f0:f0+32, c*128 : c*128+128] (B from LDS) ----
        const bf16_t* hrow = &sH[g * 16 + ln16][0];
        v8bf alo = *reinterpret_cast<const v8bf*>(hrow + half * 16);
        v8bf ahi = *reinterpret_cast<const v8bf*>(hrow + half * 16 + 8);
        v16bf afr = __builtin_shufflevector(alo, ahi, 0,1,2,3,4,5,6,7,8,9,10,11,12,13,14,15);

        const bf16_t* u2l = &sU2[cur][c * 128 + ln16][0];
#pragma unroll
        for (int dt = 0; dt < 8; ++dt) {
            v16bf bfr = load_bf16x16(u2l + (size_t)dt * 16 * 40 + half * 16);
            acc[dt] = __builtin_amdgcn_wmma_f32_16x16x32_bf16(false, afr, false, bfr,
                                                              (short)0, acc[dt], false, false);
        }
        __syncthreads();   // protect sH and sU2[cur] before they are overwritten
    }

    // ---- finalize: +b2, layernorm over d (256, split across wave pair), gelu, skip add ----
#pragma unroll
    for (int dt = 0; dt < 8; ++dt) {
        float bias = b2[br * VOCAB + c * 128 + dt * 16 + ln16];
#pragma unroll
        for (int r = 0; r < 8; ++r) acc[dt][r] += bias;
    }

    // per-wave partial sums over this wave's 128 columns
#pragma unroll
    for (int r = 0; r < 8; ++r) {
        float s = 0.f, s2 = 0.f;
#pragma unroll
        for (int dt = 0; dt < 8; ++dt) { float v = acc[dt][r]; s += v; s2 += v * v; }
        // reduce across the 16 lanes of this half (wave32: masks 1,2,4,8 stay in-half)
        s  += __shfl_xor(s, 1, 32);  s2 += __shfl_xor(s2, 1, 32);
        s  += __shfl_xor(s, 2, 32);  s2 += __shfl_xor(s2, 2, 32);
        s  += __shfl_xor(s, 4, 32);  s2 += __shfl_xor(s2, 4, 32);
        s  += __shfl_xor(s, 8, 32);  s2 += __shfl_xor(s2, 8, 32);
        if (ln16 == 0) {
            sLN[c][g * 16 + half * 8 + r][0] = s;
            sLN[c][g * 16 + half * 8 + r][1] = s2;
        }
    }
    __syncthreads();

    float mu[8], rs[8];
#pragma unroll
    for (int r = 0; r < 8; ++r) {
        int rl = g * 16 + half * 8 + r;
        float S  = sLN[0][rl][0] + sLN[1][rl][0];
        float S2 = sLN[0][rl][1] + sLN[1][rl][1];
        float m = S * (1.f / VOCAB);
        float var = S2 * (1.f / VOCAB) - m * m;
        mu[r] = m;
        rs[r] = rsqrtf(var + LN_EPS);
    }

#pragma unroll
    for (int dt = 0; dt < 8; ++dt) {
        int d = c * 128 + dt * 16 + ln16;
        float w = lnw[d], bb = lnb[d];
#pragma unroll
        for (int r = 0; r < 8; ++r) {
            float v = (acc[dt][r] - mu[r]) * rs[r] * w + bb;
            v = gelu_exact(v);
            int m = m0 + g * 16 + half * 8 + r;
            size_t off = (size_t)m * (NBR * VOCAB) + br * VOCAB + d;
            out[off] = x[off] + v;
        }
    }
}

extern "C" void kernel_launch(void* const* d_in, const int* in_sizes, int n_in,
                              void* d_out, int out_size, void* d_ws, size_t ws_size,
                              hipStream_t stream) {
    const float* x   = (const float*)d_in[0];
    const float* U1  = (const float*)d_in[1];
    const float* b1  = (const float*)d_in[2];
    const float* U2  = (const float*)d_in[3];
    const float* b2  = (const float*)d_in[4];
    const float* lnw = (const float*)d_in[5];
    const float* lnb = (const float*)d_in[6];
    float* out = (float*)d_out;

    bf16_t* U1T = (bf16_t*)d_ws;                                   // 4 MB
    bf16_t* U2T = (bf16_t*)((char*)d_ws + (size_t)NBR * FFN * VOCAB * sizeof(bf16_t)); // +4 MB

    const int w1 = NBR * FFN * VOCAB;   // 2,097,152 elements each
    cvt_u1<<<w1 / 256, 256, 0, stream>>>(U1, U1T);
    cvt_u2<<<w1 / 256, 256, 0, stream>>>(U2, U2T);

    dim3 grid(BATCH / MT, NBR);
    fused_branch_mlp<<<grid, 256, 0, stream>>>(x, U1T, b1, U2T, b2, lnw, lnb, out);
}
